// DNF_9156870275111
// MI455X (gfx1250) — compile-verified, matrix operand order checked by reading, not run on hardware
//
#include <hip/hip_runtime.h>
#include <hip/hip_bf16.h>
#include <math.h>

typedef __attribute__((ext_vector_type(16))) _Float16 v16h;
typedef __attribute__((ext_vector_type(8)))  float    v8f;

#define B_   16
#define O_   24
#define P_   552      // 24*23 ordered pairs, grouped o0-major (itertools order)
#define P0_  32
#define P1_  64
#define P2_  64
#define NK_  288      // 32 + 2*64 + 2*64
#define RC_  48       // R*C
#define C_   16
#define KC_  96       // k-chunk staged in LDS
#define PT_  16       // p rows per workgroup
#define NPT_ 35       // ceil(552/16)

// workspace layout (in floats)
#define WS_ALPHA 0
#define WS_BETA  (RC_*NK_)            // 13824
#define WS_OK    (2*RC_*NK_)          // 27648
#define WS_CONJ  (2*RC_*NK_ + RC_)    // 27696;  conj = [B*P][3][16] = 423936 floats

// ---------------------------------------------------------------------------
// Kernel 1: hoist softmax(and_kernel/T) -> (alpha, beta), sigmoid(or_kernel/T)
//   factor = x*s0 + (1-x)*s1 + s2 = alpha*x + beta,  alpha=s0-s1, beta=s1+s2
// ---------------------------------------------------------------------------
__global__ __launch_bounds__(256) void dnf_prep(const float* __restrict__ ak,
                                                const float* __restrict__ ork,
                                                const float* __restrict__ temp,
                                                float* __restrict__ ws) {
  int t = blockIdx.x * 256 + threadIdx.x;
  float invT = 1.0f / temp[0];
  if (t < RC_ * NK_) {
    float a0 = ak[t*3+0] * invT, a1 = ak[t*3+1] * invT, a2 = ak[t*3+2] * invT;
    float m  = fmaxf(a0, fmaxf(a1, a2));
    float e0 = __expf(a0 - m), e1 = __expf(a1 - m), e2 = __expf(a2 - m);
    float r  = 1.0f / (e0 + e1 + e2);
    ws[WS_ALPHA + t] = (e0 - e1) * r;
    ws[WS_BETA  + t] = (e1 + e2) * r;
  }
  if (t < RC_) ws[WS_OK + t] = 1.0f / (1.0f + __expf(-ork[t] * invT));
}

// ---------------------------------------------------------------------------
// Kernel 2: conjuncts(b,p,r,c) = prod_k (alpha*x + beta)
// One WG per (b, 16-p tile): x tile + (alpha,beta) chunk staged in LDS,
// 3 independent product accumulators per thread (breaks the serial chain;
// compiler lowers to v_pk_fma_f32 + dual-issue on wave32).
// ---------------------------------------------------------------------------
__global__ __launch_bounds__(256) void dnf_conj(const float* __restrict__ nullary,
                                                const float* __restrict__ unary,
                                                const float* __restrict__ binary,
                                                const float* __restrict__ ws,
                                                float* __restrict__ conj) {
  __shared__ float  xs[PT_][NK_];        // 18.4 KB
  __shared__ float2 ab[RC_][KC_ + 1];    // 37.2 KB (pad row -> avoid bank conflicts)
  const float* alpha = ws + WS_ALPHA;
  const float* beta  = ws + WS_BETA;

  const int b   = blockIdx.x / NPT_;
  const int pt  = blockIdx.x % NPT_;
  const int tid = threadIdx.x;

  // ---- stage x tile: rebuild permutation gather on the fly ----
  for (int i = tid; i < PT_ * NK_; i += 256) {
    int pl = i / NK_, k = i % NK_;
    int p  = pt * PT_ + pl;
    float x = 0.0f;
    if (p < P_) {
      int o0  = p / (O_ - 1);
      int rem = p % (O_ - 1);
      int o1  = rem + (rem >= o0 ? 1 : 0);
      if (k < P0_) {
        x = nullary[b * P0_ + k];
      } else if (k < P0_ + 2 * P1_) {
        int v = (k - P0_) / P1_, kk = (k - P0_) % P1_;
        int o = v ? o1 : o0;
        x = unary[(b * O_ + o) * P1_ + kk];
      } else {
        int v = (k - P0_ - 2 * P1_) / P2_, kk = (k - P0_ - 2 * P1_) % P2_;
        int i0 = v ? o1 : o0;
        int j0 = v ? (o0 - (o0 > o1 ? 1 : 0)) : rem;   // adjusted second index
        x = binary[((b * O_ + i0) * (O_ - 1) + j0) * P2_ + kk];
      }
    }
    xs[pl][k] = x;
  }

  // ---- 3 outputs per thread (16p x 48rc = 768 = 3*256) ----
  const int oi0 = tid, oi1 = tid + 256, oi2 = tid + 512;
  const int pl0 = oi0 / RC_, rc0 = oi0 % RC_;
  const int pl1 = oi1 / RC_, rc1 = oi1 % RC_;
  const int pl2 = oi2 / RC_, rc2 = oi2 % RC_;
  float pr0 = 1.0f, pr1 = 1.0f, pr2 = 1.0f;

  for (int kc = 0; kc < NK_; kc += KC_) {
    __syncthreads();                       // previous chunk fully consumed
    for (int i = tid; i < RC_ * KC_; i += 256) {
      int rc = i / KC_, kk = i % KC_;
      ab[rc][kk] = make_float2(alpha[rc * NK_ + kc + kk],
                               beta [rc * NK_ + kc + kk]);
    }
    if (kc + KC_ < NK_ && tid < RC_) {     // gfx1250 prefetch of next chunk
      __builtin_prefetch(&alpha[tid * NK_ + kc + KC_], 0, 1);
      __builtin_prefetch(&beta [tid * NK_ + kc + KC_], 0, 1);
    }
    __syncthreads();
#pragma unroll 4
    for (int kk = 0; kk < KC_; ++kk) {
      int k = kc + kk;
      float2 c0 = ab[rc0][kk]; pr0 *= fmaf(c0.x, xs[pl0][k], c0.y);
      float2 c1 = ab[rc1][kk]; pr1 *= fmaf(c1.x, xs[pl1][k], c1.y);
      float2 c2 = ab[rc2][kk]; pr2 *= fmaf(c2.x, xs[pl2][k], c2.y);
    }
  }

  const int pbase = pt * PT_;
  if (pbase + pl0 < P_) conj[((b*P_ + pbase+pl0)*3 + rc0/C_)*C_ + rc0%C_] = pr0;
  if (pbase + pl1 < P_) conj[((b*P_ + pbase+pl1)*3 + rc1/C_)*C_ + rc1%C_] = pr1;
  if (pbase + pl2 < P_) conj[((b*P_ + pbase+pl2)*3 + rc2/C_)*C_ + rc2%C_] = pr2;
}

// ---------------------------------------------------------------------------
// Kernel 3: binary_out(b,i,j') = 1 - prod_c (1 - conj(..,r=2,c)*ok2[c])
// Row-sum over C=16 in log2 space done on the matrix pipe:
//   A (16x32 f16) holds log2 values in K=0..15 (zero-padded), B = ones
//   -> v_wmma_f32_16x16x32_f16 yields S[row] in every column; lanes 0/16
//      read rows j / j+8 from accumulator VGPR j per the C/D layout.
// One wave per 16-row tile; 8832 rows = 552 tiles exactly.
// okv points at ws+WS_OK; r=2 row of sigmoid(or_kernel) is okv[32..47].
// ---------------------------------------------------------------------------
__global__ __launch_bounds__(32) void dnf_binary(const float* __restrict__ conj,
                                                 const float* __restrict__ okv,
                                                 float* __restrict__ out) {
  const int tile = blockIdx.x;           // 0..551
  const int lane = threadIdx.x;          // 0..31
  const int m    = lane & 15;            // A-matrix row
  const int ch   = (lane >> 4) * 8;      // this half-wave supplies c = ch..ch+7
  const int row  = tile * 16 + m;        // == b*552 + p == flat (b,i,j') index

  v16h a;
#pragma unroll
  for (int h = 0; h < 8; ++h) {          // K = 0..7 (lanes<16) / 8..15 (lanes>=16)
    int c = ch + h;
    float t = 1.0f - conj[(row * 3 + 2) * C_ + c] * okv[32 + c];
    a[h] = (_Float16)__log2f(t);
  }
#pragma unroll
  for (int h = 8; h < 16; ++h) a[h] = (_Float16)0.0f;  // K >= 16 padding

  v16h bm;
#pragma unroll
  for (int h = 0; h < 16; ++h) bm[h] = (_Float16)1.0f;

  v8f acc = {};
  acc = __builtin_amdgcn_wmma_f32_16x16x32_f16(false, a, false, bm,
                                               (short)0, acc, false, false);

  if (m == 0) {                          // lane 0 -> rows 0..7, lane 16 -> rows 8..15
    int rbase = 16 + 384 + tile * 16 + (lane >> 4) * 8;
#pragma unroll
    for (int j = 0; j < 8; ++j) out[rbase + j] = 1.0f - exp2f(acc[j]);
  }
}

// ---------------------------------------------------------------------------
// Kernel 4: unary_out(b,o) = 1 - prod_c (1 - U(b,o,c)*ok1[c]),
//           U = 1 - prod over the 23 perms sharing o0 = o (r=1 plane)
// ---------------------------------------------------------------------------
__global__ __launch_bounds__(128) void dnf_unary(const float* __restrict__ conj,
                                                 const float* __restrict__ okv,
                                                 float* __restrict__ out) {
  int t = blockIdx.x * 128 + threadIdx.x;
  if (t >= B_ * O_) return;
  int b = t / O_, o = t % O_;
  float acc = 1.0f;
#pragma unroll
  for (int c = 0; c < C_; ++c) {
    float pr = 1.0f;
    for (int q = 0; q < O_ - 1; ++q) {
      int row = b * P_ + o * (O_ - 1) + q;
      pr *= 1.0f - conj[(row * 3 + 1) * C_ + c];
    }
    acc *= 1.0f - (1.0f - pr) * okv[16 + c];
  }
  out[16 + t] = 1.0f - acc;
}

// ---------------------------------------------------------------------------
// Kernel 5: nullary_out(b): probsum over all 552 perms (r=0), then over c.
// One wave32 per b; cross-lane product over c via shfl_xor.
// ---------------------------------------------------------------------------
__global__ __launch_bounds__(32) void dnf_nullary(const float* __restrict__ conj,
                                                  const float* __restrict__ okv,
                                                  float* __restrict__ out) {
  int b = blockIdx.x, lane = threadIdx.x;
  float t = 1.0f;
  if (lane < C_) {
    float pr = 1.0f;
    for (int p = 0; p < P_; ++p)
      pr *= 1.0f - conj[((b * P_ + p) * 3 + 0) * C_ + lane];
    t = 1.0f - (1.0f - pr) * okv[lane];
  }
#pragma unroll
  for (int off = 16; off >= 1; off >>= 1) t *= __shfl_xor(t, off, 32);
  if (lane == 0) out[b] = 1.0f - t;
}

// ---------------------------------------------------------------------------
extern "C" void kernel_launch(void* const* d_in, const int* in_sizes, int n_in,
                              void* d_out, int out_size, void* d_ws, size_t ws_size,
                              hipStream_t stream) {
  const float* nullary = (const float*)d_in[0];
  const float* unary   = (const float*)d_in[1];
  const float* binary  = (const float*)d_in[2];
  const float* ak      = (const float*)d_in[3];
  const float* ork     = (const float*)d_in[4];
  const float* temp    = (const float*)d_in[5];
  float* out = (float*)d_out;
  float* ws  = (float*)d_ws;
  float* conj = ws + WS_CONJ;
  const float* okv = ws + WS_OK;

  dnf_prep<<<(RC_ * NK_ + 255) / 256, 256, 0, stream>>>(ak, ork, temp, ws);
  dnf_conj<<<B_ * NPT_, 256, 0, stream>>>(nullary, unary, binary, ws, conj);
  dnf_binary<<<(B_ * P_) / 16, 32, 0, stream>>>(conj, okv, out);
  dnf_unary<<<(B_ * O_ + 127) / 128, 128, 0, stream>>>(conj, okv, out);
  dnf_nullary<<<B_, 32, 0, stream>>>(conj, okv, out);
}
